// GraphAttentionLayer_31379031065006
// MI455X (gfx1250) — compile-verified, matrix operand order checked by reading, not run on hardware
//
#include <hip/hip_runtime.h>
#include <hip/hip_bf16.h>

#define N_NODES 8192
#define DIM 256
#define ROWS_PER_WG 32
#define KC 128         // j-chunk (contraction) per staging round
#define KPW 136        // W-tile row stride in halves (272 B, 16B-aligned)

typedef __attribute__((ext_vector_type(16))) _Float16 v16h;
typedef __attribute__((ext_vector_type(8)))  _Float16 v8h;
typedef __attribute__((ext_vector_type(8)))  float    v8f;

union F16Frag { v16h v; v8h h[2]; };

// ---------------- Kernel 0: hT = f16(h^T)  [DIM][N_NODES] ------------------
__global__ __launch_bounds__(256) void gat_pack_kernel(const float* __restrict__ h,
                                                       _Float16* __restrict__ hT) {
  __shared__ __align__(16) _Float16 T[64 * 72];
  const int tid = threadIdx.x;
  const int j0 = blockIdx.x * 64;   // 128 tiles along j
  const int d0 = blockIdx.y * 64;   // 4 tiles along d
  const int jj  = tid >> 4;         // 0..15
  const int dd4 = (tid & 15) * 4;   // 0..60
#pragma unroll
  for (int it = 0; it < 4; ++it) {
    const int j = jj + it * 16;
    const float4 hv = *(const float4*)(h + (size_t)(j0 + j) * DIM + d0 + dd4);
    T[(dd4 + 0) * 72 + j] = (_Float16)hv.x;
    T[(dd4 + 1) * 72 + j] = (_Float16)hv.y;
    T[(dd4 + 2) * 72 + j] = (_Float16)hv.z;
    T[(dd4 + 3) * 72 + j] = (_Float16)hv.w;
  }
  __syncthreads();
  const int dd = tid >> 2;          // 0..63
  const int js = (tid & 3) * 16;    // 0,16,32,48
  const v8h a0 = *(const v8h*)(T + dd * 72 + js);
  const v8h a1 = *(const v8h*)(T + dd * 72 + js + 8);
  _Float16* op = hT + (size_t)(d0 + dd) * N_NODES + j0 + js;
  *(v8h*)(op)     = a0;
  *(v8h*)(op + 8) = a1;
}

// ---------------- Kernel 1: h1[i] = h[i,:]·a[:D], h2[i] = h[i,:]·a[D:] ----
__global__ __launch_bounds__(256) void gat_h1h2_kernel(const float* __restrict__ h,
                                                       const float* __restrict__ a,
                                                       float* __restrict__ h1,
                                                       float* __restrict__ h2) {
  const int lane = threadIdx.x & 31;
  const int wave = threadIdx.x >> 5;
  const int row  = blockIdx.x * 8 + wave;
  const float* hr = h + (size_t)row * DIM;
  float s1 = 0.f, s2 = 0.f;
#pragma unroll
  for (int k = 0; k < 2; ++k) {
    const int d = lane * 4 + k * 128;
    const float4 hv = *(const float4*)(hr + d);
    const float4 a1 = *(const float4*)(a + d);
    const float4 a2 = *(const float4*)(a + DIM + d);
    s1 += hv.x * a1.x + hv.y * a1.y + hv.z * a1.z + hv.w * a1.w;
    s2 += hv.x * a2.x + hv.y * a2.y + hv.z * a2.z + hv.w * a2.w;
  }
#pragma unroll
  for (int off = 16; off > 0; off >>= 1) {
    s1 += __shfl_xor(s1, off, 32);
    s2 += __shfl_xor(s2, off, 32);
  }
  if (lane == 0) { h1[row] = s1; h2[row] = s2; }
}

// ---------------- Kernel 2: global max of h2 (softmax stabilizer bound) ----
__global__ __launch_bounds__(256) void gat_max_kernel(const float* __restrict__ h2,
                                                      float* __restrict__ mx) {
  __shared__ float red[256];
  float m = -3.4e38f;
  for (int j = threadIdx.x; j < N_NODES; j += 256) m = fmaxf(m, h2[j]);
  red[threadIdx.x] = m;
  __syncthreads();
  for (int s = 128; s > 0; s >>= 1) {
    if (threadIdx.x < s) red[threadIdx.x] = fmaxf(red[threadIdx.x], red[threadIdx.x + s]);
    __syncthreads();
  }
  if (threadIdx.x == 0) *mx = red[0];
}

// ---------------- Kernel 3: fused masked-softmax + (attn @ h) via WMMA -----
__global__ __launch_bounds__(512) void gat_attn_kernel(const _Float16* __restrict__ hT,
                                                       const int*   __restrict__ adj,
                                                       const float* __restrict__ h1,
                                                       const float* __restrict__ h2,
                                                       const float* __restrict__ maxp,
                                                       float* __restrict__ out) {
  __shared__ __align__(16) _Float16 Wt[ROWS_PER_WG * KPW]; // weights tile [32][KC] f16
  __shared__ float denom_s[ROWS_PER_WG];
  __shared__ float h1s[ROWS_PER_WG];
  __shared__ float ms[ROWS_PER_WG];

  const int tid  = threadIdx.x;
  const int lane = tid & 31;
  const int wave = tid >> 5;       // 0..15 -> d-block
  const int d0   = wave * 16;
  const int hh   = lane >> 4;      // half-wave select (K groups / C row groups)
  const int nm   = lane & 15;      // M for A frags, N for B frags
  const int i0   = blockIdx.x * ROWS_PER_WG;
  const float maxh2 = *maxp;

  if (tid < ROWS_PER_WG) {
    const float v1 = h1[i0 + tid];
    h1s[tid] = v1;
    const float s = v1 + maxh2;                // upper bound of row max (lrelu monotonic)
    ms[tid] = (s > 0.f) ? s : 0.01f * s;
    denom_s[tid] = 0.f;
  }

  v8f acc[2] = {};
  // Static staging map: thread owns W-tile row (tid>>4), 8 cols per chunk.
  const int wrow = tid >> 4;          // 0..31
  const int wc0  = (tid & 15) * 8;    // 0..120
  const int irow = i0 + wrow;
  const int* adjr = adj + (size_t)irow * N_NODES;
  const _Float16* bbase = hT + (size_t)(d0 + nm) * N_NODES;
  float dpart = 0.f;

  for (int j0 = 0; j0 < N_NODES; j0 += KC) {
    __syncthreads();   // prior round's fragment reads complete
    // ---- stage W tile: w = adj ? exp(lrelu(h1+h2) - m) : 0, packed f16 ----
    {
      const int4   ad0 = *(const int4*)  (adjr + j0 + wc0);
      const int4   ad1 = *(const int4*)  (adjr + j0 + wc0 + 4);
      const float4 h2a = *(const float4*)(h2 + j0 + wc0);
      const float4 h2b = *(const float4*)(h2 + j0 + wc0 + 4);
      const float hb = h1s[wrow];
      const float mb = ms[wrow];
      const float sv[8]  = {hb + h2a.x, hb + h2a.y, hb + h2a.z, hb + h2a.w,
                            hb + h2b.x, hb + h2b.y, hb + h2b.z, hb + h2b.w};
      const int   adv[8] = {ad0.x, ad0.y, ad0.z, ad0.w, ad1.x, ad1.y, ad1.z, ad1.w};
      v8h pk;
#pragma unroll
      for (int q = 0; q < 8; ++q) {
        const float e = sv[q] > 0.f ? sv[q] : 0.01f * sv[q];
        const float w = (adv[q] != 0) ? __expf(e - mb) : 0.f;
        const _Float16 wq = (_Float16)w;
        pk[q] = wq;
        dpart += (float)wq;   // denominator from same quantized value: consistent
      }
      *(v8h*)(Wt + wrow * KPW + wc0) = pk;   // single ds_store_b128
    }
    __syncthreads();
    // ---- WMMA: four K=32 steps; B fragments straight from global hT -------
#pragma unroll
    for (int s = 0; s < 4; ++s) {
      const int kb = s * 32 + hh * 8;        // ISA 16-bit layout: VGPR0-3 = K[hh*8+0..7]
      F16Frag bf;
      const _Float16* bp = bbase + j0 + kb;
      bf.h[0] = *(const v8h*)(bp);           // global_load_b128
      bf.h[1] = *(const v8h*)(bp + 16);      // VGPR4-7 = K[16+hh*8+0..7]
#pragma unroll
      for (int ib = 0; ib < 2; ++ib) {
        F16Frag af;
        const _Float16* ap = Wt + (ib * 16 + nm) * KPW + kb;
        af.h[0] = *(const v8h*)(ap);
        af.h[1] = *(const v8h*)(ap + 16);
        acc[ib] = __builtin_amdgcn_wmma_f32_16x16x32_f16(
            false, af.v, false, bf.v, (short)0, acc[ib], false, false);
      }
    }
  }

  // ---- denominators: reduce 16 partials per row, then scale + store -------
  atomicAdd(&denom_s[wrow], dpart);
  __syncthreads();

#pragma unroll
  for (int ib = 0; ib < 2; ++ib) {
#pragma unroll
    for (int r = 0; r < 8; ++r) {
      const int ml = ib * 16 + hh * 8 + r;   // C/D layout: VGPR r, half hh -> row
      const float dn  = denom_s[ml];
      const float inv = dn > 0.f ? __builtin_amdgcn_rcpf(dn) : 0.f;
      out[(size_t)(i0 + ml) * DIM + d0 + nm] = acc[ib][r] * inv;
    }
  }
}

extern "C" void kernel_launch(void* const* d_in, const int* in_sizes, int n_in,
                              void* d_out, int out_size, void* d_ws, size_t ws_size,
                              hipStream_t stream) {
  const float* h   = (const float*)d_in[0];
  const int*   adj = (const int*)  d_in[1];
  const float* a   = (const float*)d_in[2];
  float* ws = (float*)d_ws;
  float* h1 = ws;                              // [8192]
  float* h2 = ws + N_NODES;                    // [8192]
  float* mx = ws + 2 * N_NODES;                // [1]
  _Float16* hT = (_Float16*)(ws + 16640);      // [256][8192] f16, 16B-aligned
  float* out = (float*)d_out;

  gat_pack_kernel<<<dim3(N_NODES / 64, DIM / 64), 256, 0, stream>>>(h, hT);
  gat_h1h2_kernel<<<N_NODES / 8, 256, 0, stream>>>(h, a, h1, h2);
  gat_max_kernel<<<1, 256, 0, stream>>>(h2, mx);
  gat_attn_kernel<<<N_NODES / ROWS_PER_WG, 512, 0, stream>>>(hT, adj, h1, h2, mx, out);
}
